// MultiModalityFusion_85066122264761
// MI455X (gfx1250) — matrix operand order repaired and checked by reading, not compiled, not run on hardware
//
#include <hip/hip_runtime.h>
#include <hip/hip_bf16.h>

// ---------------------------------------------------------------------------
// MultiModalityFusion fused implementation for gfx1250 (MI455X, wave32, WMMA)
// B matrices stored in LDS pre-swizzled into WMMA fragment layout so each
// fragment load is 2x ds_load_b128 (no u16 gathers / b16 packing movs).
// ---------------------------------------------------------------------------

#define BB   8
#define CC   256
#define HH   128
#define WW   128
#define HWSZ 16384          // H*W
#define MCH  6
#define CBN  64             // SE bottleneck
#define EPSF 1e-5f
#define TPB  256
#define TILE 32             // pixels per tile (one row segment)

#define AP   264            // aT: halves per pixel-row (padded)
#define AMP  72             // aTm: halves per pixel-row (K=64 padded)
#define FP   260            // fT / mT: floats per pixel-row

typedef __attribute__((ext_vector_type(16))) _Float16 v16h;
typedef __attribute__((ext_vector_type(8)))  _Float16 h8;
typedef __attribute__((ext_vector_type(8)))  float    v8f;

// ---- LDS layout for the fused kernel (all offsets 16B aligned) -------------
#define OFF_BT    0                          // 256*256 halves = 131072 B (swizzled)
#define OFF_BTM   (OFF_BT   + 256*256*2)     //  64*256 halves =  32768 B (swizzled)
#define OFF_DW    (OFF_BTM  + 64*256*2)      // 2304 f32       =   9216 B
#define OFF_MC2   (OFF_DW   + 2304*4)        // 2304 f32       =   9216 B
#define OFF_MC1   (OFF_MC2  + 2304*4)        //  128 f32       =    512 B
#define OFF_FNW   (OFF_MC1  + 128*4)         //  256 f32
#define OFF_MNW   (OFF_FNW  + 256*4)
#define OFF_KNW   (OFF_MNW  + 256*4)
#define OFF_SSQ   (OFF_KNW  + 256*4)         //   64 f32 (ssqM[32], ssqK[32])
#define OFF_FT    (OFF_SSQ  + 64*4)          // 32*FP f32 = 33280 B (aliases aT)
#define OFF_MT    (OFF_FT   + 32*FP*4)       // 32*FP f32 = 33280 B
#define OFF_ATM   (OFF_MT   + 32*FP*4)       // 32*AMP halves = 4608 B
#define SMEM_TOTAL (OFF_ATM + 32*AMP*2)      // = 257280 B  (<= 320KB WGP LDS)

// fragment-swizzled destination index for B element (k, n):
//   kb = k>>5, lane = (n&15) + ((k>>4)&1)*16, half = k&15, ntile = n>>4
__device__ __forceinline__ int bswz(int k, int n) {
    int kb   = k >> 5;
    int lane = (n & 15) + (((k >> 4) & 1) << 4);
    int half = k & 15;
    int ntg  = n >> 4;
    return ((((kb << 4) + ntg) << 5) + lane) * 16 + half;
}

// ---- monotone float<->uint keys so max-pool can use u32 atomicMax ----------
__device__ __forceinline__ unsigned fkey(float f) {
    unsigned u = __float_as_uint(f);
    return (u & 0x80000000u) ? ~u : (u | 0x80000000u);
}
__device__ __forceinline__ float fdec(unsigned u) {
    return (u & 0x80000000u) ? __uint_as_float(u ^ 0x80000000u)
                             : __uint_as_float(~u);
}
__device__ __forceinline__ float silu(float s) {
    return s * (1.0f / (1.0f + __expf(-s)));
}

// ===========================================================================
// Kernel 0: init pool keys (ws is poisoned by harness; max needs identity)
// ===========================================================================
__global__ void k_init_pool(unsigned* __restrict__ pool) {
    int i = blockIdx.x * TPB + threadIdx.x;
    if (i < BB * CC) pool[i] = 0u;
}

// ===========================================================================
// Kernel 1: per-pixel inverse RMS over channels + channel-wise spatial max of
//           x = img * inv_rms * feat_norm_w   (the SE "adaptive max pool")
// ===========================================================================
__global__ void k_prep(const float* __restrict__ img,
                       const float* __restrict__ fnw,
                       float* __restrict__ invr,
                       unsigned* __restrict__ pool) {
    int pix = blockIdx.x * TPB + threadIdx.x;      // 131072 pixels
    int b   = pix >> 14;
    int pi  = pix & (HWSZ - 1);
    const float* ip = img + ((long)b << 22) + pi;  // b*C*HW + pi, channel stride HW

    float ssq = 0.f;
    for (int c = 0; c < CC; ++c) {
        float v = ip[c << 14];
        ssq += v * v;
    }
    float inv = rsqrtf(ssq * (1.f / 256.f) + EPSF);
    invr[pix] = inv;

    // spatial max per channel: wave-reduce (32 consecutive pixels, same b)
    for (int c = 0; c < CC; ++c) {
        float v = ip[c << 14] * inv * fnw[c];
        for (int s = 16; s > 0; s >>= 1)
            v = fmaxf(v, __shfl_xor(v, s, 32));
        if ((threadIdx.x & 31) == 0)
            atomicMax(&pool[(b << 8) + c], fkey(v));
    }
}

// ===========================================================================
// Kernel 2: SE MLP  gate[b,c] = W2 @ relu(rmsnorm(W1 @ pool[b]))
// ===========================================================================
__global__ void k_se(const unsigned* __restrict__ pool,
                     const float* __restrict__ w1,
                     const float* __restrict__ senw,
                     const float* __restrict__ w2,
                     float* __restrict__ gate) {
    __shared__ float pl[CC];
    __shared__ float red[CBN];
    __shared__ float h[CBN];
    int b = blockIdx.x, tid = threadIdx.x;

    pl[tid] = fdec(pool[(b << 8) + tid]);
    __syncthreads();

    float v = 0.f;
    if (tid < CBN) {
        const float* wr = w1 + tid * CC;
        for (int c = 0; c < CC; ++c) v += wr[c] * pl[c];
        red[tid] = v * v;
    }
    __syncthreads();
    for (int s = 32; s > 0; s >>= 1) {
        if (tid < s) red[tid] += red[tid + s];
        __syncthreads();
    }
    float inv = rsqrtf(red[0] * (1.f / 64.f) + EPSF);
    if (tid < CBN) h[tid] = fmaxf(v * inv * senw[tid], 0.f);
    __syncthreads();

    float g = 0.f;
    const float* w2r = w2 + tid * CBN;
    for (int j = 0; j < CBN; ++j) g += w2r[j] * h[j];
    gate[(b << 8) + tid] = g;
}

// ===========================================================================
// Kernel 3: fused main pass (persistent blocks, 32-pixel row tiles)
// ===========================================================================
__global__ void __launch_bounds__(TPB, 1)
k_fused(const float* __restrict__ img,
        const float* __restrict__ modals,
        const float* __restrict__ mask,
        const float* __restrict__ fnw_g,
        const float* __restrict__ dw_g,
        const float* __restrict__ pw_g,
        const float* __restrict__ mw_g,
        const float* __restrict__ mnw_g,
        const float* __restrict__ mc1_g,
        const float* __restrict__ mc2_g,
        const float* __restrict__ knw_g,
        const float* __restrict__ invr_g,
        const float* __restrict__ gate_g,
        float* __restrict__ out,
        int ntiles) {
    extern __shared__ char smem[];
    _Float16* bT   = (_Float16*)(smem + OFF_BT);   // pw_w  fragments [kb=8][ntg=16][lane=32][16]
    _Float16* bTm  = (_Float16*)(smem + OFF_BTM);  // modal fragments [kb=2][ntg=16][lane=32][16]
    float*    dwW  = (float*)(smem + OFF_DW);
    float*    mc2  = (float*)(smem + OFF_MC2);
    float*    mc1  = (float*)(smem + OFF_MC1);
    float*    fnw  = (float*)(smem + OFF_FNW);
    float*    mnw  = (float*)(smem + OFF_MNW);
    float*    knw  = (float*)(smem + OFF_KNW);
    float*    ssqM = (float*)(smem + OFF_SSQ);
    float*    ssqK = ssqM + 32;
    float*    fT   = (float*)(smem + OFF_FT);      // GEMM-1 output tile
    float*    mT   = (float*)(smem + OFF_MT);      // GEMM-2 output tile
    _Float16* aTm  = (_Float16*)(smem + OFF_ATM);  // modal A tile [32][64]
    _Float16* aT   = (_Float16*)fT;                // depthwise A tile (aliases fT)

    const int tid  = threadIdx.x;
    const int wave = tid >> 5, lane = tid & 31;
    const int p    = tid >> 3, cl   = tid & 7;     // pixel-in-tile / channel lane
    const int mt   = wave & 1;                      // M tile (16 rows)
    const int wn   = wave >> 1;                     // N strip (64 cols)
    const int hsel = lane >> 4;                     // lane half

    // ---- stage weights once per block (f32 -> f16, fragment swizzle) ------
    for (int i = tid; i < CC * CC; i += TPB) {      // pw_g[cout][cin]: n=cout, k=cin
        int n = i >> 8, k = i & 255;
        bT[bswz(k, n)] = (_Float16)pw_g[i];
    }
    for (int i = tid; i < 64 * CC; i += TPB) {      // modal_w[c][kk]: n=c, k=kk (pad to 64)
        int k = i >> 8, n = i & 255;
        bTm[bswz(k, n)] = (_Float16)((k < 54) ? mw_g[n * 54 + k] : 0.f);
    }
    for (int i = tid; i < 2304; i += TPB) { dwW[i] = dw_g[i]; mc2[i] = mc2_g[i]; }
    if (tid < 128) mc1[tid] = mc1_g[tid];
    fnw[tid] = fnw_g[tid]; mnw[tid] = mnw_g[tid]; knw[tid] = knw_g[tid];
    __syncthreads();

    for (int t = blockIdx.x; t < ntiles; t += gridDim.x) {
        const int base = t << 5;
        const int b    = base >> 14;
        const int pi   = base & (HWSZ - 1);
        const int y    = pi >> 7;
        const int x    = (pi & 127) + p;
        const long pixbase = ((long)(b << 8) << 14) + ((long)y << 7) + x;

        if (t + (int)gridDim.x < ntiles) {          // global_prefetch_b8 next tile
            int nb = (t + gridDim.x) << 5;
            __builtin_prefetch(img + (((long)((nb >> 14) << 8)) << 14) + (nb & (HWSZ - 1)), 0, 0);
        }
        if (tid < 32) { ssqM[tid] = 0.f; ssqK[tid] = 0.f; }

        // ---- per-pixel 3x3 inverse-RMS neighborhood (0 outside image) -----
        float invr9[9];
        #pragma unroll
        for (int dy = 0; dy < 3; ++dy)
            #pragma unroll
            for (int dx = 0; dx < 3; ++dx) {
                int yy = y + dy - 1, xx = x + dx - 1;
                invr9[dy * 3 + dx] =
                    (yy >= 0 && yy < HH && xx >= 0 && xx < WW)
                        ? invr_g[(b << 14) + (yy << 7) + xx] : 0.f;
            }
        const float ir = invr9[4];

        // ---- phase 1a: depthwise 3x3 on x = img*inv_rms*fnw -> relu -> aT --
        #pragma unroll 4
        for (int j = 0; j < 32; ++j) {
            int c = cl + (j << 3);
            const float* ip = img + ((long)((b << 8) + c) << 14);
            const float* dw = dwW + c * 9;
            float acc = 0.f;
            #pragma unroll
            for (int dy = 0; dy < 3; ++dy) {
                int yy = y + dy - 1;
                #pragma unroll
                for (int dx = 0; dx < 3; ++dx) {
                    int xx = x + dx - 1;
                    float r = invr9[dy * 3 + dx];
                    if (r != 0.f)
                        acc += dw[dy * 3 + dx] * ip[(yy << 7) + xx] * r;
                }
            }
            aT[p * AP + c] = (_Float16)fmaxf(acc * fnw[c], 0.f);
        }
        // ---- phase 1b: modal 3x3 patch -> aTm (K padded to 64) ------------
        #pragma unroll
        for (int q = 0; q < 8; ++q) {
            int kk = cl + (q << 3);
            float v = 0.f;
            if (kk < 54) {
                int ic = kk / 9, k9 = kk % 9;
                int yy = y + k9 / 3 - 1, xx = x + k9 % 3 - 1;
                if (yy >= 0 && yy < HH && xx >= 0 && xx < WW)
                    v = modals[((long)(b * MCH + ic) << 14) + (yy << 7) + xx];
            }
            aTm[p * AMP + kk] = (_Float16)v;
        }
        __syncthreads();

        // ---- phase 2: two WMMA GEMMs (fragments load as 2x ds_load_b128) --
        v8f acc0 = {}, acc1 = {}, acc2 = {}, acc3 = {};
        v8f mac0 = {}, mac1 = {}, mac2 = {}, mac3 = {};
        const int row = mt * 16 + (lane & 15);

        #pragma unroll
        for (int kb = 0; kb < 8; ++kb) {            // main GEMM: K=256
            union { v16h v; struct { h8 lo, hi; } s; } ua;
            const _Float16* ar = aT + row * AP + kb * 32 + hsel * 16;
            ua.s.lo = *(const h8*)(ar);
            ua.s.hi = *(const h8*)(ar + 8);
            v16h av = ua.v;
            #pragma unroll
            for (int nt = 0; nt < 4; ++nt) {
                const _Float16* fb =
                    bT + (((((kb << 4) + (wn << 2) + nt) << 5) + lane) << 4);
                union { v16h v; struct { h8 lo, hi; } s; } ub;
                ub.s.lo = *(const h8*)(fb);
                ub.s.hi = *(const h8*)(fb + 8);
                v8f* ap = (nt == 0) ? &acc0 : (nt == 1) ? &acc1 : (nt == 2) ? &acc2 : &acc3;
                *ap = __builtin_amdgcn_wmma_f32_16x16x32_f16(
                          false, av, false, ub.v, (short)0, *ap, false, false);
            }
        }
        #pragma unroll
        for (int kb = 0; kb < 2; ++kb) {            // modal GEMM: K=64
            union { v16h v; struct { h8 lo, hi; } s; } ua;
            const _Float16* ar = aTm + row * AMP + kb * 32 + hsel * 16;
            ua.s.lo = *(const h8*)(ar);
            ua.s.hi = *(const h8*)(ar + 8);
            v16h av = ua.v;
            #pragma unroll
            for (int nt = 0; nt < 4; ++nt) {
                const _Float16* fb =
                    bTm + (((((kb << 4) + (wn << 2) + nt) << 5) + lane) << 4);
                union { v16h v; struct { h8 lo, hi; } s; } ub;
                ub.s.lo = *(const h8*)(fb);
                ub.s.hi = *(const h8*)(fb + 8);
                v8f* ap = (nt == 0) ? &mac0 : (nt == 1) ? &mac1 : (nt == 2) ? &mac2 : &mac3;
                *ap = __builtin_amdgcn_wmma_f32_16x16x32_f16(
                          false, av, false, ub.v, (short)0, *ap, false, false);
            }
        }
        __syncthreads();                            // all aT reads done (fT aliases aT)

        // ---- D tiles -> LDS (ISA D layout: vgpr r -> M=r(+8), N=lane&15) --
        #pragma unroll
        for (int nt = 0; nt < 4; ++nt) {
            v8f a  = (nt == 0) ? acc0 : (nt == 1) ? acc1 : (nt == 2) ? acc2 : acc3;
            v8f ma = (nt == 0) ? mac0 : (nt == 1) ? mac1 : (nt == 2) ? mac2 : mac3;
            #pragma unroll
            for (int r = 0; r < 8; ++r) {
                int m = mt * 16 + r + hsel * 8;
                int n = (wn << 6) + (nt << 4) + (lane & 15);
                fT[m * FP + n] = a[r];
                mT[m * FP + n] = ma[r];
            }
        }
        __syncthreads();

        // ---- phase 3: mask branch + per-pixel sum-of-squares --------------
        float mp[9];
        #pragma unroll
        for (int k9 = 0; k9 < 9; ++k9) {
            int yy = y + k9 / 3 - 1, xx = x + k9 % 3 - 1;
            mp[k9] = (yy >= 0 && yy < HH && xx >= 0 && xx < WW)
                         ? mask[((long)b << 14) + (yy << 7) + xx] : 0.f;
        }
        float mk[32];
        float sm = 0.f, sk = 0.f;
        #pragma unroll 4
        for (int j = 0; j < 32; ++j) {
            int c = cl + (j << 3);
            const float* kc = mc2 + c * 9;
            float ak = 0.f;
            #pragma unroll
            for (int q = 0; q < 9; ++q) ak += kc[q] * mp[q];
            ak *= mc1[c >> 1];
            mk[j] = ak;
            sk += ak * ak;
            float mv = mT[p * FP + c];
            sm += mv * mv;
        }
        atomicAdd(&ssqM[p], sm);
        atomicAdd(&ssqK[p], sk);
        __syncthreads();
        if (tid < 32) {
            ssqM[tid] = rsqrtf(ssqM[tid] * (1.f / 256.f) + EPSF);
            ssqK[tid] = rsqrtf(ssqK[tid] * (1.f / 256.f) + EPSF);
        }
        __syncthreads();

        // ---- phase 4: gates + combine + store -----------------------------
        const float im = ssqM[p], ik = ssqK[p];
        #pragma unroll 4
        for (int j = 0; j < 32; ++j) {
            int  c  = cl + (j << 3);
            long gi = pixbase + ((long)c << 14);
            float xv = img[gi] * ir * fnw[c];
            float fm = silu(gate_g[(b << 8) + c] * xv);      // feat_mul
            float fv = fT[p * FP + c] * fm;                  // f branch
            float mv = mT[p * FP + c] * im * mnw[c] * fm;    // modal branch
            float mg = silu(mk[j] * ik * knw[c]);            // mask gate
            out[gi] = (fv + mv) * (1.f + mg);
        }
        __syncthreads();                                     // before fT/aT reuse
    }
}

// ===========================================================================
extern "C" void kernel_launch(void* const* d_in, const int* in_sizes, int n_in,
                              void* d_out, int out_size, void* d_ws, size_t ws_size,
                              hipStream_t stream) {
    (void)in_sizes; (void)n_in; (void)out_size; (void)ws_size;
    const float* img    = (const float*)d_in[0];
    const float* modals = (const float*)d_in[1];
    const float* mask   = (const float*)d_in[2];
    const float* fnw    = (const float*)d_in[3];
    const float* dw_w   = (const float*)d_in[4];
    const float* pw_w   = (const float*)d_in[5];
    const float* mw_w   = (const float*)d_in[6];
    const float* mnw    = (const float*)d_in[7];
    const float* se_w1  = (const float*)d_in[8];
    const float* senw   = (const float*)d_in[9];
    const float* se_w2  = (const float*)d_in[10];
    const float* mc1    = (const float*)d_in[11];
    const float* mc2    = (const float*)d_in[12];
    const float* knw    = (const float*)d_in[13];
    float* out = (float*)d_out;

    char* ws = (char*)d_ws;
    float*    invr = (float*)ws;                           // 131072 f32 = 512KB
    unsigned* pool = (unsigned*)(ws + 131072 * 4);         // 2048 u32
    float*    gate = (float*)(ws + 131072 * 4 + 2048 * 4); // 2048 f32

    const int npix   = BB * HWSZ;          // 131072
    const int ntiles = npix / TILE;        // 4096

    k_init_pool<<<(BB * CC + TPB - 1) / TPB, TPB, 0, stream>>>(pool);
    k_prep<<<npix / TPB, TPB, 0, stream>>>(img, fnw, invr, pool);
    k_se<<<BB, TPB, 0, stream>>>(pool, se_w1, senw, se_w2, gate);
    k_fused<<<1024, TPB, SMEM_TOTAL, stream>>>(
        img, modals, mask, fnw, dw_w, pw_w, mw_w, mnw,
        mc1, mc2, knw, invr, gate, out, ntiles);
}
